// AtomToPair_10505490006485
// MI455X (gfx1250) — compile-verified, matrix operand order checked by reading, not run on hardware
//
#include <hip/hip_runtime.h>

typedef __attribute__((ext_vector_type(16))) _Float16 v16h;
typedef __attribute__((ext_vector_type(8)))  float    v8f;

#define NB 32
#define NA 64
#define NF 128
#define NC 128
#define NROWS (NB*NA)            // 2048 atom rows
#define NPAIR (NB*NA*NA)         // 131072 pair rows
#define NSTRIP (NPAIR/16)        // 8192 strips of 16 pairs
#define WAVES_PER_BLOCK 8

// ---------------------------------------------------------------------------
// Prep 1: U' = X @ W0[0:128,:] + b0   and   V = X @ W0[128:256,:]
// One thread per (b,a,c). Tiny (67M MACs total), W0 is L2-resident.
// ---------------------------------------------------------------------------
__global__ void precompute_uv(const float* __restrict__ x,
                              const float* __restrict__ W0,
                              const float* __restrict__ b0,
                              float* __restrict__ Up,
                              float* __restrict__ Vp) {
    int idx = blockIdx.x * blockDim.x + threadIdx.x;   // [0, 32*64*128)
    int c   = idx & (NC - 1);
    int row = idx >> 7;                                // b*64 + a
    const float* xr = x + (row << 7);
    float aU = b0[c];
    float aV = 0.0f;
#pragma unroll 4
    for (int k = 0; k < NF; ++k) {
        float xv = xr[k];
        aU = fmaf(xv, W0[k * NC + c], aU);
        aV = fmaf(xv, W0[(k + NF) * NC + c], aV);
    }
    Up[idx] = aU;
    Vp[idx] = aV;
}

// ---------------------------------------------------------------------------
// Prep 2: repack W1 (128x128 f32, row-major W1[k][n]) into f16 B-fragments
// laid out exactly as V_WMMA_F32_16X16X32_F16 wants for lane-direct loads:
//   Bpack[nt][ks][lane][e],  k = ks*32 + (lane>>4)*8 + (e<8 ? e : 16+(e-8)),
//   n = nt*16 + (lane&15).   8 ntiles * 4 ksteps * 32 lanes * 16 elems = 16384.
// ---------------------------------------------------------------------------
__global__ void pack_w1(const float* __restrict__ W1, _Float16* __restrict__ W1p) {
    int t = blockIdx.x * blockDim.x + threadIdx.x;     // [0, 16384)
    int e    = t & 15;
    int lane = (t >> 4) & 31;
    int ks   = (t >> 9) & 3;
    int nt   = t >> 11;
    int k = ks * 32 + ((lane >> 4) * 8) + (e < 8 ? e : 8 + e);  // e>=8 -> +16+(e-8)
    int n = nt * 16 + (lane & 15);
    W1p[t] = (_Float16)W1[k * NC + n];
}

// ---------------------------------------------------------------------------
// Main: per wave, one strip of 16 pairs (fixed b,i ; j = j0..j0+15) x 128 out.
// A0 = relu(U'[i] + V[j])   (h0 path),  A1 = relu(U'[j] + V[i])  (h1 path),
// built straight into WMMA A-fragment registers. W1 fragments from LDS.
// out[b, i*64+j, :] = relu(A0@W1 + b1) + relu(A1@W1 + b1)
// ---------------------------------------------------------------------------
__global__ __launch_bounds__(256) void pair_mlp2(const float* __restrict__ Up,
                                                 const float* __restrict__ Vp,
                                                 const _Float16* __restrict__ W1p,
                                                 const float* __restrict__ b1,
                                                 float* __restrict__ out) {
    __shared__ alignas(32) _Float16 sW[8 * 4 * 32 * 16];   // 32 KB

    // cooperative stage of packed W1 into LDS (2048 x 16B)
    {
        const uint4* src = (const uint4*)W1p;
        uint4*       dst = (uint4*)sW;
        for (int t = threadIdx.x; t < 2048; t += 256) dst[t] = src[t];
    }
    __syncthreads();

    const int wave = threadIdx.x >> 5;
    const int lane = threadIdx.x & 31;
    const int s    = blockIdx.x * WAVES_PER_BLOCK + wave;   // strip id, < 8192
    const int p0   = s << 4;                                // first pair row
    const int b    = p0 >> 12;
    const int i    = (p0 >> 6) & (NA - 1);
    const int j0   = p0 & (NA - 1);                         // multiple of 16

    const int m     = lane & 15;     // A-matrix row this lane contributes
    const int khalf = lane >> 4;     // which K-half this lane holds

    const float* Ui = Up + (size_t)((b * NA + i) << 7);
    const float* Vi = Vp + (size_t)((b * NA + i) << 7);
    const float* Uj = Up + (size_t)((b * NA + j0 + m) << 7);
    const float* Vj = Vp + (size_t)((b * NA + j0 + m) << 7);

    // Build A fragments: 4 k-steps of 16x32 f16 for each of the two paths.
    v16h A0[4], A1[4];
#pragma unroll
    for (int ks = 0; ks < 4; ++ks) {
        const int kb = ks * 32 + khalf * 8;
#pragma unroll
        for (int seg = 0; seg < 2; ++seg) {
            const int kk = kb + seg * 16;
#pragma unroll
            for (int t = 0; t < 8; ++t) {
                float h0 = Ui[kk + t] + Vj[kk + t];     // row0 = relu(U'[i]+V[j])
                float h1 = Uj[kk + t] + Vi[kk + t];     // row1 = relu(U'[j]+V[i])
                h0 = fmaxf(h0, 0.0f);
                h1 = fmaxf(h1, 0.0f);
                A0[ks][seg * 8 + t] = (_Float16)h0;
                A1[ks][seg * 8 + t] = (_Float16)h1;
            }
        }
    }

    const int   ncol      = lane & 15;
    const int   prow_base = p0 + khalf * 8;

#pragma unroll
    for (int nt = 0; nt < 8; ++nt) {
        v8f acc0 = {};
        v8f acc1 = {};
#pragma unroll
        for (int ks = 0; ks < 4; ++ks) {
            v16h bf = *(const v16h*)(sW + (((nt * 4 + ks) * 32) + lane) * 16);
            acc0 = __builtin_amdgcn_wmma_f32_16x16x32_f16(
                false, A0[ks], false, bf, (short)0, acc0, false, false);
            acc1 = __builtin_amdgcn_wmma_f32_16x16x32_f16(
                false, A1[ks], false, bf, (short)0, acc1, false, false);
        }
        const float bias = b1[nt * 16 + ncol];
        const int   n    = nt * 16 + ncol;
#pragma unroll
        for (int r = 0; r < 8; ++r) {
            float h0 = fmaxf(acc0[r] + bias, 0.0f);
            float h1 = fmaxf(acc1[r] + bias, 0.0f);
            out[(size_t)(prow_base + r) * NC + n] = h0 + h1;
        }
    }
}

// ---------------------------------------------------------------------------
extern "C" void kernel_launch(void* const* d_in, const int* in_sizes, int n_in,
                              void* d_out, int out_size, void* d_ws, size_t ws_size,
                              hipStream_t stream) {
    const float* x  = (const float*)d_in[0];   // (32,64,128)
    const float* W0 = (const float*)d_in[1];   // (256,128)
    const float* b0 = (const float*)d_in[2];   // (128,)
    const float* W1 = (const float*)d_in[3];   // (128,128)
    const float* b1 = (const float*)d_in[4];   // (128,)
    float* out = (float*)d_out;                // (32,4096,128)

    // workspace layout: Up (1MB) | Vp (1MB) | W1 packed f16 (32KB)
    float*     Up  = (float*)d_ws;
    float*     Vp  = Up + (size_t)NROWS * NC;
    _Float16*  W1p = (_Float16*)(Vp + (size_t)NROWS * NC);

    precompute_uv<<<(NROWS * NC) / 256, 256, 0, stream>>>(x, W0, b0, Up, Vp);
    pack_w1<<<16384 / 256, 256, 0, stream>>>(W1, W1p);
    pair_mlp2<<<NSTRIP / WAVES_PER_BLOCK, 256, 0, stream>>>(Up, Vp, W1p, b1, out);
}
